// SNNLayer_32985348834028
// MI455X (gfx1250) — matrix operand order repaired and checked by reading, not compiled, not run on hardware
//
#include <hip/hip_runtime.h>
#include <hip/hip_bf16.h>
#include <stdint.h>

// SNN on MI455X (gfx1250, wave32, WMMA).
// Pipeline:
//   1) pack_split: w1,w2 f32 -> f16 hi/lo pairs (split-f16 ~= f32 GEMM accuracy)
//   2) gemm1_spikebits: x@w1^T+b1 via v_wmma_f32_16x16x32_f16 (3-term hi/lo
//      split product). Each wave converts its A tile once per k-step and feeds
//      12 WMMAs (4 col tiles) to amortize the f32->f16 VALU conversions.
//      Epilogue simulates the 8 LIF steps exactly (reset-to-zero semantics,
//      same repeated-add FP order as the reference) and stores an 8-bit spike
//      schedule per neuron (bit t == spike at step t). No f32 intermediate.
//   3) snn_steps: for t=0..7, mem2 tiles live in registers across timesteps as
//      WMMA C-operands; A = spikes rebuilt from the bitmask (exact 0/1 in f16),
//      B = w2 hi/lo streamed from L2 (2 MB, L2-resident). Threshold+store+reset
//      per t. No LDS required anywhere (A operand is 16 bytes/lane/k-step).

typedef __attribute__((ext_vector_type(16))) _Float16 v16h;
typedef __attribute__((ext_vector_type(8)))  float    v8f;

#define WMMA16(A, Bm, C) \
  __builtin_amdgcn_wmma_f32_16x16x32_f16(false, (A), false, (Bm), (short)0, (C), false, false)

#define B_  8
#define S_  1024
#define I_  512
#define H_  1024
#define O_  512
#define T_  8
#define THRESH_ 0.5f

// ---------------------------------------------------------------- pack weights
__global__ __launch_bounds__(256) void pack_split(const float* __restrict__ w,
                                                  _Float16* __restrict__ hi,
                                                  _Float16* __restrict__ lo,
                                                  int n) {
  int i = blockIdx.x * 256 + threadIdx.x;
  if (i < n) {
    float v = w[i];
    _Float16 h = (_Float16)v;
    hi[i] = h;
    lo[i] = (_Float16)(v - (float)h);
  }
}

// --------------------------------------------- layer 1 GEMM + LIF spike schedule
// grid: 1024 blocks (128 row-blocks x 8 col-blocks of 64x128), 256 threads
// (8 waves). Wave: 1 row-tile x 4 col-tiles -> one A split feeds 12 WMMAs.
__global__ __launch_bounds__(256) void gemm1_spikebits(
    const float* __restrict__ x, const _Float16* __restrict__ w1h,
    const _Float16* __restrict__ w1l, const float* __restrict__ b1,
    uint8_t* __restrict__ bits) {
  const int lane = threadIdx.x & 31;
  const int wave = threadIdx.x >> 5;
  const int m16  = lane & 15;  // A row / B-C column within tile
  const int kh   = lane >> 4;  // K-half selector
  const int rb = blockIdx.x >> 3;   // 0..127 (64 rows each)
  const int cb = blockIdx.x & 7;    // 0..7   (128 cols each)
  const int rowtile = wave & 3;
  const int ct0     = (wave >> 2) * 4;  // 4 col tiles per wave

  const int arow = rb * 64 + rowtile * 16 + m16;
  const int col0 = cb * 128 + ct0 * 16;

  const float* __restrict__ xrow = x + (size_t)arow * I_;

  v8f acc[4];
#pragma unroll
  for (int ct = 0; ct < 4; ++ct) acc[ct] = (v8f){};

  for (int kt = 0; kt < I_; kt += 32) {
    __builtin_prefetch(xrow + kt + 64, 0, 0);  // global_prefetch_b8
    // A operand: f32 -> f16 hi/lo, contiguous K layout (elements 0..15 = K
    // 16*kh+0..15, matching B)
    float af[16];
    *(float4*)(af + 0)  = *(const float4*)(xrow + kt + 16 * kh + 0);
    *(float4*)(af + 4)  = *(const float4*)(xrow + kt + 16 * kh + 4);
    *(float4*)(af + 8)  = *(const float4*)(xrow + kt + 16 * kh + 8);
    *(float4*)(af + 12) = *(const float4*)(xrow + kt + 16 * kh + 12);
    v16h Ah, Al;
#pragma unroll
    for (int e = 0; e < 16; ++e) {
      _Float16 h = (_Float16)af[e];
      Ah[e] = h;
      Al[e] = (_Float16)(af[e] - (float)h);
    }
    // B operand: w1 is (H,I) row-major == [N][K], K contiguous per lane.
#pragma unroll
    for (int ct = 0; ct < 4; ++ct) {
      const size_t bo = (size_t)(col0 + ct * 16 + m16) * I_ + kt + 16 * kh;
      v16h Bh = *(const v16h*)(w1h + bo);
      v16h Bl = *(const v16h*)(w1l + bo);
      acc[ct] = WMMA16(Ah, Bh, acc[ct]);
      acc[ct] = WMMA16(Ah, Bl, acc[ct]);
      acc[ct] = WMMA16(Al, Bh, acc[ct]);
    }
  }

  // Epilogue: add b1, simulate the 8 LIF steps exactly (reset-to-zero), store
  // spike schedule bitmask. C layout: M = v + 8*kh, N = lane&15.
  const int crow = rb * 64 + rowtile * 16 + 8 * kh;
#pragma unroll
  for (int ct = 0; ct < 4; ++ct) {
    const int n = col0 + ct * 16 + m16;
    const float bv = b1[n];
#pragma unroll
    for (int v = 0; v < 8; ++v) {
      const int r = crow + v;
      float c = acc[ct][v] + bv;
      float m = 0.f;
      uint32_t bt = 0;
#pragma unroll
      for (int tt = 0; tt < T_; ++tt) {
        m += c;
        if (m > THRESH_) { bt |= 1u << tt; m = 0.f; }
      }
      bits[(size_t)r * H_ + n] = (uint8_t)bt;
    }
  }
}

// ------------------------------------------------- layer 2: T timesteps, fused
// grid: 128 blocks x 512 threads (16 waves). Block = 64 rows x all 512 cols.
// Wave: rowtile = wave>>2 (4 of them), 8 col tiles -> 8 register-resident
// mem2 accumulators carried across all T timesteps (WMMA C-operand chaining).
__global__ __launch_bounds__(512) void snn_steps(
    const uint8_t* __restrict__ bits, const _Float16* __restrict__ w2h,
    const _Float16* __restrict__ w2l, const float* __restrict__ b2,
    float* __restrict__ out) {
  const int lane = threadIdx.x & 31;
  const int wave = threadIdx.x >> 5;  // 0..15
  const int m16  = lane & 15;
  const int kh   = lane >> 4;
  const int rowbase = blockIdx.x * 64 + (wave >> 2) * 16;
  const int ctg     = (wave & 3) * 8;  // first of this wave's 8 col tiles

  v8f mem2[8];
#pragma unroll
  for (int ct = 0; ct < 8; ++ct) mem2[ct] = (v8f){};
  float b2v[8];
#pragma unroll
  for (int ct = 0; ct < 8; ++ct) b2v[ct] = b2[(ctg + ct) * 16 + m16];

  const uint8_t* __restrict__ prow = bits + (size_t)(rowbase + m16) * H_;

  for (int t = 0; t < T_; ++t) {
    for (int kt = 0; kt < H_; kt += 32) {
      __builtin_prefetch(prow + kt + 32, 0, 0);
      // A = spike1_t rebuilt from schedule bits (exact 0/1 in f16)
      const uint2 p0 = *(const uint2*)(prow + kt + 16 * kh);
      const uint2 p1 = *(const uint2*)(prow + kt + 16 * kh + 8);
      v16h A;
#pragma unroll
      for (int e = 0; e < 16; ++e) {
        uint32_t w = (e < 4) ? p0.x : (e < 8) ? p0.y : (e < 12) ? p1.x : p1.y;
        uint32_t byte = (w >> ((e & 3) * 8)) & 0xFFu;
        A[e] = (_Float16)(float)((byte >> t) & 1u);
      }
#pragma unroll
      for (int ct = 0; ct < 8; ++ct) {
        const size_t boff = (size_t)((ctg + ct) * 16 + m16) * H_ + kt + 16 * kh;
        v16h Bh = *(const v16h*)(w2h + boff);
        v16h Bl = *(const v16h*)(w2l + boff);
        mem2[ct] = WMMA16(A, Bh, mem2[ct]);
        mem2[ct] = WMMA16(A, Bl, mem2[ct]);
      }
    }
    // per-timestep epilogue: mem2 += b2; spike2 out; reset fired neurons
#pragma unroll
    for (int ct = 0; ct < 8; ++ct) {
      const int o = (ctg + ct) * 16 + m16;
#pragma unroll
      for (int v = 0; v < 8; ++v) {
        const int r = rowbase + 8 * kh + v;       // global row = b*S + s
        float val = mem2[ct][v] + b2v[ct];
        const bool fire = val > THRESH_;
        const size_t oi =
            ((size_t)((r >> 10) * T_ + t) * S_ + (size_t)(r & 1023)) * O_ + o;
        out[oi] = fire ? 1.0f : 0.0f;
        mem2[ct][v] = fire ? 0.0f : val;
      }
    }
  }
}

// ---------------------------------------------------------------------- launch
extern "C" void kernel_launch(void* const* d_in, const int* in_sizes, int n_in,
                              void* d_out, int out_size, void* d_ws, size_t ws_size,
                              hipStream_t stream) {
  const float* x  = (const float*)d_in[0];
  const float* w1 = (const float*)d_in[1];
  const float* b1 = (const float*)d_in[2];
  const float* w2 = (const float*)d_in[3];
  const float* b2 = (const float*)d_in[4];
  float* out = (float*)d_out;

  char* ws = (char*)d_ws;
  const size_t MB = 1u << 20;
  _Float16* w1h = (_Float16*)(ws + 0 * MB);  // 1 MB (1024x512 f16)
  _Float16* w1l = (_Float16*)(ws + 1 * MB);  // 1 MB
  _Float16* w2h = (_Float16*)(ws + 2 * MB);  // 1 MB (512x1024 f16)
  _Float16* w2l = (_Float16*)(ws + 3 * MB);  // 1 MB
  uint8_t*  bits = (uint8_t*)(ws + 4 * MB);  // 8 MB spike schedules

  pack_split<<<(H_ * I_ + 255) / 256, 256, 0, stream>>>(w1, w1h, w1l, H_ * I_);
  pack_split<<<(O_ * H_ + 255) / 256, 256, 0, stream>>>(w2, w2h, w2l, O_ * H_);
  gemm1_spikebits<<<1024, 256, 0, stream>>>(x, w1h, w1l, b1, bits);
  snn_steps<<<128, 512, 0, stream>>>(bits, w2h, w2l, b2, out);
}
